// GNN_Bet_67688684585008
// MI455X (gfx1250) — compile-verified, matrix operand order checked by reading, not run on hardware
//
#include <hip/hip_runtime.h>

// ---------------------------------------------------------------------------
// GNN_Bet on MI455X (gfx1250).
//   - All dense GEMMs: bf16 operands, f32 accumulate, V_WMMA_F32_16X16X32_BF16,
//     2x global_load_b128 per fragment (A and B pre-laid-out / pre-transposed).
//   - SpMM: edge-parallel, bf16 gather (halves L2 traffic), f32 atomic scatter.
// ---------------------------------------------------------------------------

typedef __attribute__((ext_vector_type(16))) __bf16 v16bf;
typedef __attribute__((ext_vector_type(8)))  __bf16 bf16x8;
typedef __attribute__((ext_vector_type(4)))  __bf16 bf16x4;
typedef __attribute__((ext_vector_type(8)))  float  v8f;

union frag16 { v16bf v; bf16x8 h[2]; };

__device__ __forceinline__ __bf16 f2bf(float f) {
  unsigned u = __builtin_bit_cast(unsigned, f);
  unsigned r = (u + 0x7FFFu + ((u >> 16) & 1u)) >> 16;   // round-to-nearest-even
  return __builtin_bit_cast(__bf16, (unsigned short)r);
}
__device__ __forceinline__ float bf2f(__bf16 b) {
  unsigned u = ((unsigned)__builtin_bit_cast(unsigned short, b)) << 16;
  return __builtin_bit_cast(float, u);
}

// ---------------------------------------------------------------------------
// C[M x Nc](bf16) = act( A[M x K](bf16) @ BT^T + bias ),
// BT is [Nc x K] row-major (B already transposed), f32 accumulation.
// Requires M % 16 == 0, K % 32 == 0, Nc % 128 == 0.
// Block = 256 thr = 8 waves; block tile = 16 rows x 128 cols, wave = 16x16.
// ---------------------------------------------------------------------------
__global__ void __launch_bounds__(256)
gemm_bf16_wmma_kernel(const __bf16* __restrict__ A, const __bf16* __restrict__ BT,
                      __bf16* __restrict__ C, int M, int K, int Nc,
                      const float* __restrict__ bias, int relu)
{
  const int wave = threadIdx.x >> 5;
  const int lane = threadIdx.x & 31;
  const int half = lane >> 4;               // 0 or 1
  const int l16  = lane & 15;
  const int row0 = blockIdx.y * 16;
  const int col0 = blockIdx.x * 128 + wave * 16;

  // ISA 16-bit A 16x32 layout: lanes 0-15 hold K {0..7,16..23}, lanes 16-31
  // hold K {8..15,24..31}. Two contiguous 8-element (16B) runs per fragment.
  const __bf16* arow = A  + (size_t)(row0 + l16) * K + half * 8;
  const __bf16* brow = BT + (size_t)(col0 + l16) * K + half * 8;

  v8f acc = {};
  for (int kk = 0; kk < K; kk += 32) {
    frag16 a, b;
    a.h[0] = *reinterpret_cast<const bf16x8*>(arow + kk);
    a.h[1] = *reinterpret_cast<const bf16x8*>(arow + kk + 16);
    b.h[0] = *reinterpret_cast<const bf16x8*>(brow + kk);
    b.h[1] = *reinterpret_cast<const bf16x8*>(brow + kk + 16);
    acc = __builtin_amdgcn_wmma_f32_16x16x32_bf16(
        /*neg_a=*/false, a.v, /*neg_b=*/false, b.v,
        /*c_mod=*/(short)0, acc, /*reuse_a=*/false, /*reuse_b=*/false);
  }

  // f32 C/D layout: lane<16 -> col=lane, elem r = row r; lane>=16 -> row r+8.
  const int ncol  = col0 + l16;
  const int crow0 = row0 + half * 8;
  const float bb = bias ? bias[ncol] : 0.0f;
#pragma unroll
  for (int r = 0; r < 8; ++r) {
    float v = acc[r] + bb;
    if (relu) v = fmaxf(v, 0.0f);
    C[(size_t)(crow0 + r) * Nc + ncol] = f2bf(v);
  }
}

// ---------------------------------------------------------------------------
// AGG[row] += val * X[col]  (H = 128), 32 lanes per edge, 4 bf16 gathers each.
// ---------------------------------------------------------------------------
__global__ void __launch_bounds__(256)
spmm_kernel(const int* __restrict__ row, const int* __restrict__ col,
            const float* __restrict__ val, const __bf16* __restrict__ X,
            float* __restrict__ AGG, int E)
{
  const int gid  = blockIdx.x * blockDim.x + threadIdx.x;
  const int e    = gid >> 5;
  if (e >= E) return;
  const int lane = gid & 31;
  const int c = col[e];
  const int r = row[e];
  const float v = val[e];
  const bf16x4 x = *reinterpret_cast<const bf16x4*>(X + (size_t)c * 128 + lane * 4);
  float* y = AGG + (size_t)r * 128 + lane * 4;
  atomicAdd(y + 0, v * bf2f(x[0]));
  atomicAdd(y + 1, v * bf2f(x[1]));
  atomicAdd(y + 2, v * bf2f(x[2]));
  atomicAdd(y + 3, v * bf2f(x[3]));
}

__global__ void __launch_bounds__(256)
zero_kernel(float* __restrict__ p, size_t n)
{
  size_t i = (size_t)blockIdx.x * blockDim.x + threadIdx.x;
  if (i < n) p[i] = 0.0f;
}

__global__ void __launch_bounds__(256)
f32_to_bf16_kernel(const float* __restrict__ in, __bf16* __restrict__ out, size_t n)
{
  size_t i = (size_t)blockIdx.x * blockDim.x + threadIdx.x;
  if (i < n) out[i] = f2bf(in[i]);
}

// out[c * R + r] = in[r * Cc + c]   (small weight transpose, f32 -> bf16)
__global__ void __launch_bounds__(256)
transpose_f32_to_bf16_kernel(const float* __restrict__ in, __bf16* __restrict__ out,
                             int R, int Cc)
{
  int i = blockIdx.x * blockDim.x + threadIdx.x;
  if (i >= R * Cc) return;
  int r = i / Cc, c = i % Cc;
  out[(size_t)c * R + r] = f2bf(in[(size_t)r * Cc + c]);
}

// ---------------------------------------------------------------------------
// Xb[node] = maybe_l2norm( relu( AGG[node] + bias ) )  -> bf16, wave per node.
// ---------------------------------------------------------------------------
__global__ void __launch_bounds__(256)
bias_act_norm_kernel(const float* __restrict__ AGG, const float* __restrict__ bias,
                     __bf16* __restrict__ Xb, int n_nodes, int do_norm)
{
  const int gid  = blockIdx.x * blockDim.x + threadIdx.x;
  const int node = gid >> 5;
  if (node >= n_nodes) return;
  const int lane = gid & 31;

  float4 y = *reinterpret_cast<const float4*>(AGG + (size_t)node * 128 + lane * 4);
  const float4 b = *reinterpret_cast<const float4*>(bias + lane * 4);
  y.x = fmaxf(y.x + b.x, 0.0f);
  y.y = fmaxf(y.y + b.y, 0.0f);
  y.z = fmaxf(y.z + b.z, 0.0f);
  y.w = fmaxf(y.w + b.w, 0.0f);

  if (do_norm) {
    float ss = y.x * y.x + y.y * y.y + y.z * y.z + y.w * y.w;
#pragma unroll
    for (int off = 16; off >= 1; off >>= 1) ss += __shfl_xor(ss, off, 32);
    const float inv = 1.0f / fmaxf(sqrtf(ss), 1e-12f);
    y.x *= inv; y.y *= inv; y.z *= inv; y.w *= inv;
  }
  bf16x4 o; o[0] = f2bf(y.x); o[1] = f2bf(y.y); o[2] = f2bf(y.z); o[3] = f2bf(y.w);
  *reinterpret_cast<bf16x4*>(Xb + (size_t)node * 128 + lane * 4) = o;
}

// ---------------------------------------------------------------------------
// S[node] += dot(H2[node, 0:256](bf16), w(f32)) + b[0], wave per node.
// ---------------------------------------------------------------------------
__global__ void __launch_bounds__(256)
score_matvec_kernel(const __bf16* __restrict__ H2, const float* __restrict__ w,
                    const float* __restrict__ b, float* __restrict__ S, int n_nodes)
{
  const int gid  = blockIdx.x * blockDim.x + threadIdx.x;
  const int node = gid >> 5;
  if (node >= n_nodes) return;
  const int lane = gid & 31;

  const __bf16* h = H2 + (size_t)node * 256;
  float s = 0.0f;
#pragma unroll
  for (int i = 0; i < 8; ++i) s += bf2f(h[lane + 32 * i]) * w[lane + 32 * i];
#pragma unroll
  for (int off = 16; off >= 1; off >>= 1) s += __shfl_xor(s, off, 32);
  if (lane == 0) S[node] += s + b[0];
}

__global__ void __launch_bounds__(256)
mul_kernel(const float* __restrict__ a, const float* __restrict__ b,
           float* __restrict__ o, int n)
{
  int i = blockIdx.x * blockDim.x + threadIdx.x;
  if (i < n) o[i] = a[i] * b[i];
}

// ---------------------------------------------------------------------------
extern "C" void kernel_launch(void* const* d_in, const int* in_sizes, int n_in,
                              void* d_out, int out_size, void* d_ws, size_t ws_size,
                              hipStream_t stream)
{
  (void)n_in; (void)ws_size; (void)out_size;
  const int N = 50000;
  const int E = in_sizes[0];

  const int*   in_row  = (const int*)  d_in[0];
  const int*   in_col  = (const int*)  d_in[1];
  const float* in_val  = (const float*)d_in[2];
  const int*   out_row = (const int*)  d_in[3];
  const int*   out_col = (const int*)  d_in[4];
  const float* out_val = (const float*)d_in[5];
  const float* W1 = (const float*)d_in[6];  const float* b1 = (const float*)d_in[7];
  const float* W2 = (const float*)d_in[8];  const float* b2 = (const float*)d_in[9];
  const float* W3 = (const float*)d_in[10]; const float* b3 = (const float*)d_in[11];
  const float* W4 = (const float*)d_in[12]; const float* b4 = (const float*)d_in[13];
  const float* lin1_w = (const float*)d_in[14]; const float* lin1_b = (const float*)d_in[15];
  const float* lin2_w = (const float*)d_in[16]; const float* lin2_b = (const float*)d_in[17];
  const float* lin3_w = (const float*)d_in[18]; const float* lin3_b = (const float*)d_in[19];

  // ---- workspace carve-up ------------------------------------------------
  char* wp = (char*)d_ws;
  auto alloc = [&](size_t bytes) -> void* {
    void* q = (void*)wp; wp += (bytes + 255) & ~(size_t)255; return q;
  };
  float*  AGG  = (float*) alloc((size_t)N * 128 * 4);
  float*  Sin  = (float*) alloc((size_t)N * 4);
  float*  Sout = (float*) alloc((size_t)N * 4);
  __bf16* Xb   = (__bf16*)alloc((size_t)N * 128 * 2);
  __bf16* Yb   = (__bf16*)alloc((size_t)N * 128 * 2);
  __bf16* H1b  = (__bf16*)alloc((size_t)N * 256 * 2);
  __bf16* H2b  = (__bf16*)alloc((size_t)N * 256 * 2);
  __bf16* W1b  = (__bf16*)alloc((size_t)N * 128 * 2);
  __bf16* W2T  = (__bf16*)alloc((size_t)128 * 128 * 2);
  __bf16* W3T  = (__bf16*)alloc((size_t)128 * 128 * 2);
  __bf16* W4T  = (__bf16*)alloc((size_t)128 * 128 * 2);
  __bf16* L1b  = (__bf16*)alloc((size_t)256 * 128 * 2);
  __bf16* L2b  = (__bf16*)alloc((size_t)256 * 256 * 2);

  const size_t nh = (size_t)N * 128;
  const dim3 blk(256);
  auto nblk = [](size_t n) { return (unsigned)((n + 255) / 256); };

  // ---- one-time operand preparation (re-run every call: deterministic) ---
  f32_to_bf16_kernel<<<nblk(nh), blk, 0, stream>>>(W1, W1b, nh);
  transpose_f32_to_bf16_kernel<<<nblk(128 * 128), blk, 0, stream>>>(W2, W2T, 128, 128);
  transpose_f32_to_bf16_kernel<<<nblk(128 * 128), blk, 0, stream>>>(W3, W3T, 128, 128);
  transpose_f32_to_bf16_kernel<<<nblk(128 * 128), blk, 0, stream>>>(W4, W4T, 128, 128);
  f32_to_bf16_kernel<<<nblk(256 * 128), blk, 0, stream>>>(lin1_w, L1b, 256 * 128);
  f32_to_bf16_kernel<<<nblk(256 * 256), blk, 0, stream>>>(lin2_w, L2b, 256 * 256);
  zero_kernel<<<nblk((size_t)N), blk, 0, stream>>>(Sin, (size_t)N);
  zero_kernel<<<nblk((size_t)N), blk, 0, stream>>>(Sout, (size_t)N);

  auto gemm = [&](const __bf16* A, const __bf16* BT, __bf16* C, int M, int K, int Nc,
                  const float* bias, int relu) {
    dim3 grid(Nc / 128, M / 16);
    gemm_bf16_wmma_kernel<<<grid, blk, 0, stream>>>(A, BT, C, M, K, Nc, bias, relu);
  };

  auto score = [&](const __bf16* x, float* S) {
    gemm(x,   L1b, H1b, N, 128, 256, lin1_b, 1);   // relu(x @ lin1_w^T + b)
    gemm(H1b, L2b, H2b, N, 256, 256, lin2_b, 1);   // relu(h @ lin2_w^T + b)
    score_matvec_kernel<<<nblk((size_t)N * 32), blk, 0, stream>>>(H2b, lin3_w, lin3_b, S, N);
  };

  auto spmm = [&](const int* row, const int* col, const float* val, const __bf16* dense) {
    zero_kernel<<<nblk(nh), blk, 0, stream>>>(AGG, nh);
    spmm_kernel<<<nblk((size_t)E * 32), blk, 0, stream>>>(row, col, val, dense, AGG, E);
  };

  auto branch = [&](const int* row, const int* col, const float* val, float* S) {
    // layer 1: x1 = l2norm(relu(spmm(A, W1) + b1))
    spmm(row, col, val, W1b);
    bias_act_norm_kernel<<<nblk((size_t)N * 32), blk, 0, stream>>>(AGG, b1, Xb, N, 1);
    score(Xb, S);
    // layers 2..4
    const __bf16* Ws[3] = {W2T, W3T, W4T};
    const float*  bs[3] = {b2, b3, b4};
    for (int l = 0; l < 3; ++l) {
      gemm(Xb, Ws[l], Yb, N, 128, 128, nullptr, 0);   // y = x @ W
      spmm(row, col, val, Yb);
      bias_act_norm_kernel<<<nblk((size_t)N * 32), blk, 0, stream>>>(
          AGG, bs[l], Xb, N, (l < 2) ? 1 : 0);        // x4 skips l2norm
      score(Xb, S);
    }
  };

  branch(in_row,  in_col,  in_val,  Sin);
  branch(out_row, out_col, out_val, Sout);

  mul_kernel<<<nblk((size_t)N), blk, 0, stream>>>(Sin, Sout, (float*)d_out, N);
}
